// MoE_45603962749526
// MI455X (gfx1250) — matrix operand order repaired and checked by reading, not compiled, not run on hardware
//
#include <hip/hip_runtime.h>

#define T_TOK 8192
#define D_DIM 1024
#define NEXP 16
#define MAX_SLOTS (T_TOK * 2 + NEXP * 16) /* 16640 */
#define MAX_TILES (MAX_SLOTS / 16)        /* 1040 */

typedef __attribute__((ext_vector_type(16))) __bf16 v16bf;
typedef __attribute__((ext_vector_type(8)))  __bf16 v8bf;
typedef __attribute__((ext_vector_type(4)))  __bf16 v4bf;
typedef __attribute__((ext_vector_type(2)))  __bf16 v2bf;
typedef __attribute__((ext_vector_type(8)))  float  v8f;

// ---- f32 -> bf16 (RNE), packed pair: prefer hardware v_cvt_pk_bf16_f32 ----
__device__ __forceinline__ __bf16 f2bf_scalar(float f) {
  unsigned u = __builtin_bit_cast(unsigned, f);
  unsigned r = u + 0x7FFFu + ((u >> 16) & 1u);
  unsigned short h = (unsigned short)(r >> 16);
  return __builtin_bit_cast(__bf16, h);
}

__device__ __forceinline__ v2bf f2bf2(float a, float b) {
#if __has_builtin(__builtin_amdgcn_cvt_pk_bf16_f32)
  return __builtin_amdgcn_cvt_pk_bf16_f32(a, b);
#else
  v2bf r; r[0] = f2bf_scalar(a); r[1] = f2bf_scalar(b); return r;
#endif
}

__device__ __forceinline__ void pack4(v16bf& r, int base, float4 f) {
  v2bf lo = f2bf2(f.x, f.y), hi = f2bf2(f.z, f.w);
  r[base + 0] = lo[0]; r[base + 1] = lo[1];
  r[base + 2] = hi[0]; r[base + 3] = hi[1];
}

__device__ __forceinline__ void pack4v(v4bf& r, float4 f) {
  v2bf lo = f2bf2(f.x, f.y), hi = f2bf2(f.z, f.w);
  r[0] = lo[0]; r[1] = lo[1]; r[2] = hi[0]; r[3] = hi[1];
}

// ---- on-the-fly f32 loaders (fallback path) ----
__device__ __forceinline__ v16bf load16_bf(const float* __restrict__ p) {
  const float4* q = (const float4*)p;
  v16bf r;
  pack4(r, 0, q[0]); pack4(r, 4, q[1]); pack4(r, 8, q[2]); pack4(r, 12, q[3]);
  return r;
}

__device__ __forceinline__ v16bf load16_bf_sum(const float* __restrict__ p0,
                                               const float* __restrict__ p1) {
  const float4* qa = (const float4*)p0;
  const float4* qb = (const float4*)p1;
  v16bf r;
#pragma unroll
  for (int i = 0; i < 4; ++i) {
    float4 a = qa[i], b = qb[i];
    float4 s; s.x = a.x + b.x; s.y = a.y + b.y; s.z = a.z + b.z; s.w = a.w + b.w;
    pack4(r, 4 * i, s);
  }
  return r;
}

__device__ __forceinline__ v16bf loadA_bf(const float* __restrict__ row, int kb, int khalf) {
  const float4* q0 = (const float4*)(row + kb + khalf * 16);
  const float4* q1 = (const float4*)(row + kb + khalf * 16 + 8);
  v16bf r;
  pack4(r, 0, q0[0]); pack4(r, 4, q0[1]); pack4(r, 8, q1[0]); pack4(r, 12, q1[1]);
  return r;
}

// ---- preconverted bf16 A loader (fast path) ----
__device__ __forceinline__ v16bf loadA_pre(const __bf16* __restrict__ row, int kb, int khalf) {
  v8bf lo = *(const v8bf*)(row + kb + khalf * 16);
  v8bf hi = *(const v8bf*)(row + kb + khalf * 16 + 8);
  v16bf r;
#pragma unroll
  for (int i = 0; i < 8; ++i) { r[i] = lo[i]; r[8 + i] = hi[i]; }
  return r;
}

// ---------------- 1. workspace init ----------------
__global__ void moe_init_kernel(int* cnt, int* cursor, float* pi_sum,
                                int* tok_ids, float* tok_w) {
  int gid = blockIdx.x * blockDim.x + threadIdx.x;
  if (gid < NEXP) { cnt[gid] = 0; cursor[gid] = 0; pi_sum[gid] = 0.f; }
  for (int i = gid; i < MAX_SLOTS; i += gridDim.x * blockDim.x) {
    tok_ids[i] = 0; tok_w[i] = 0.f;
  }
}

// ---------------- 2. router ----------------
__global__ __launch_bounds__(256) void router_kernel(
    const float* __restrict__ feat, const float* __restrict__ router_w,
    int* __restrict__ topk_idx, float* __restrict__ topk_wt,
    int* __restrict__ cnt, float* __restrict__ pi_sum) {
  __shared__ float lds_rw[NEXP * 256];
  __shared__ float pi_blk[NEXP];
  __shared__ int cnt_blk[NEXP];
  int tid = threadIdx.x;
  int t = blockIdx.x * 256 + tid;
  if (tid < NEXP) { pi_blk[tid] = 0.f; cnt_blk[tid] = 0; }

  float acc[NEXP];
#pragma unroll
  for (int e = 0; e < NEXP; ++e) acc[e] = 0.f;

  for (int c = 0; c < 4; ++c) {
    __syncthreads();
    for (int i = tid; i < NEXP * 256; i += 256)
      lds_rw[i] = router_w[(i >> 8) * D_DIM + c * 256 + (i & 255)];
    __syncthreads();
    const float* xr = feat + (size_t)t * D_DIM + c * 256;
    for (int dd = 0; dd < 256; ++dd) {
      float xv = xr[dd];
#pragma unroll
      for (int e = 0; e < NEXP; ++e)
        acc[e] = fmaf(xv, lds_rw[e * 256 + dd], acc[e]);
    }
  }

  float mx = acc[0];
#pragma unroll
  for (int e = 1; e < NEXP; ++e) mx = fmaxf(mx, acc[e]);
  float sc[NEXP]; float sum = 0.f;
#pragma unroll
  for (int e = 0; e < NEXP; ++e) { sc[e] = __expf(acc[e] - mx); sum += sc[e]; }
  float inv = 1.0f / sum;

  int i1 = 0; float v1 = acc[0];
#pragma unroll
  for (int e = 1; e < NEXP; ++e) if (acc[e] > v1) { v1 = acc[e]; i1 = e; }
  int i2 = -1; float v2 = -__builtin_inff();
#pragma unroll
  for (int e = 0; e < NEXP; ++e)
    if (e != i1 && acc[e] > v2) { v2 = acc[e]; i2 = e; }

  float w1 = 1.0f / (1.0f + __expf(v2 - v1));
  float w2 = 1.0f - w1;

  topk_idx[t * 2 + 0] = i1; topk_idx[t * 2 + 1] = i2;
  topk_wt[t * 2 + 0] = w1;  topk_wt[t * 2 + 1] = w2;

  atomicAdd(&cnt_blk[i1], 1);
  atomicAdd(&cnt_blk[i2], 1);
#pragma unroll
  for (int e = 0; e < NEXP; ++e) atomicAdd(&pi_blk[e], sc[e] * inv);
  __syncthreads();
  if (tid < NEXP) {
    atomicAdd(&cnt[tid], cnt_blk[tid]);
    atomicAdd(&pi_sum[tid], pi_blk[tid]);
  }
}

// ---------------- 3. scan ----------------
__global__ void scan_kernel(const int* __restrict__ cnt, const float* __restrict__ pi_sum,
                            int* __restrict__ offs, int* __restrict__ total_tiles,
                            float* __restrict__ aux_out) {
  if (threadIdx.x == 0 && blockIdx.x == 0) {
    int tot = 0; float aux = 0.f;
    for (int e = 0; e < NEXP; ++e) {
      offs[e] = tot;
      tot += (cnt[e] + 15) & ~15;
      float fi = (float)cnt[e] * (1.0f / (float)T_TOK);
      float pe = pi_sum[e] * (1.0f / (float)T_TOK);
      aux += fi * pe;
    }
    offs[NEXP] = tot;
    *total_tiles = tot >> 4;
    *aux_out = aux;
  }
}

// ---------------- 4. scatter ----------------
__global__ void scatter_kernel(const int* __restrict__ topk_idx, const float* __restrict__ topk_wt,
                               const int* __restrict__ offs, int* __restrict__ cursor,
                               int* __restrict__ tok_ids, float* __restrict__ tok_w) {
  int t = blockIdx.x * blockDim.x + threadIdx.x;
  if (t >= T_TOK) return;
#pragma unroll
  for (int k = 0; k < 2; ++k) {
    int e = topk_idx[t * 2 + k];
    int pos = atomicAdd(&cursor[e], 1);
    int slot = offs[e] + pos;
    tok_ids[slot] = t;
    tok_w[slot] = topk_wt[t * 2 + k];
  }
}

// ---------------- 5a. flat f32 -> bf16 conversion ----------------
__global__ __launch_bounds__(256) void cvt_f32_bf16_kernel(
    const float* __restrict__ src, __bf16* __restrict__ dst, int n4) {
  int i = blockIdx.x * blockDim.x + threadIdx.x;
  if (i >= n4) return;
  float4 f = ((const float4*)src)[i];
  v4bf r; pack4v(r, f);
  ((v4bf*)dst)[i] = r;
}

// ---------------- 5b. shared-weight combine + convert ----------------
__global__ __launch_bounds__(256) void cvt_shared_combine_kernel(
    const float* __restrict__ shared_w, __bf16* __restrict__ dst, int n4) {
  int i = blockIdx.x * blockDim.x + threadIdx.x;
  if (i >= n4) return;
  float4 a = ((const float4*)shared_w)[i];
  float4 b = ((const float4*)(shared_w + (size_t)D_DIM * D_DIM))[i];
  float4 s; s.x = a.x + b.x; s.y = a.y + b.y; s.z = a.z + b.z; s.w = a.w + b.w;
  v4bf r; pack4v(r, s);
  ((v4bf*)dst)[i] = r;
}

// ---------------- 6a. shared GEMM, preconverted bf16 (fast) ----------------
// block = 128 thr (4 waves); wave = 16 rows x 64 cols (4 accumulators)
__global__ __launch_bounds__(128) void shared_gemm_bf_kernel(
    const __bf16* __restrict__ xbf, const __bf16* __restrict__ wbf,
    float* __restrict__ out) {
  int tile = blockIdx.x >> 2;       // T/16 = 512 token tiles
  int colgrp = blockIdx.x & 3;      // 4 groups of 256 cols
  int lane = threadIdx.x & 31;
  int wv = threadIdx.x >> 5;
  int colbase = colgrp * 256 + wv * 64;
  int m = lane & 15, khalf = lane >> 4, n = m;

  const __bf16* arow = xbf + (size_t)(tile * 16 + m) * D_DIM;
  const __bf16* brow = wbf + (size_t)(colbase + n) * D_DIM + khalf * 16;

  v8f c0 = {}, c1 = {}, c2 = {}, c3 = {};
  for (int kb = 0; kb < D_DIM; kb += 32) {
    v16bf a = loadA_pre(arow, kb, khalf);
    v16bf b0 = *(const v16bf*)(brow + kb);
    v16bf b1 = *(const v16bf*)(brow + (size_t)16 * D_DIM + kb);
    v16bf b2 = *(const v16bf*)(brow + (size_t)32 * D_DIM + kb);
    v16bf b3 = *(const v16bf*)(brow + (size_t)48 * D_DIM + kb);
    c0 = __builtin_amdgcn_wmma_f32_16x16x32_bf16(false, a, false, b0, (short)0, c0, false, false);
    c1 = __builtin_amdgcn_wmma_f32_16x16x32_bf16(false, a, false, b1, (short)0, c1, false, false);
    c2 = __builtin_amdgcn_wmma_f32_16x16x32_bf16(false, a, false, b2, (short)0, c2, false, false);
    c3 = __builtin_amdgcn_wmma_f32_16x16x32_bf16(false, a, false, b3, (short)0, c3, false, false);
  }
#pragma unroll
  for (int j = 0; j < 8; ++j) {
    int mj = j + khalf * 8;
    float* o = out + (size_t)(tile * 16 + mj) * D_DIM + colbase + n;
    o[0] = c0[j]; o[16] = c1[j]; o[32] = c2[j]; o[48] = c3[j];
  }
}

// ---------------- 6b. grouped expert GEMM, preconverted bf16 (fast) ----------------
__global__ __launch_bounds__(128) void moe_gemm_bf_kernel(
    const __bf16* __restrict__ xbf, const __bf16* __restrict__ webf,
    const int* __restrict__ tok_ids, const float* __restrict__ tok_w,
    const int* __restrict__ offs, const int* __restrict__ total_tiles,
    float* __restrict__ out) {
  int tile = blockIdx.x >> 2;
  if (tile >= *total_tiles) return;
  int colgrp = blockIdx.x & 3;
  int slotbase = tile * 16;

  int e = 0;
#pragma unroll
  for (int k = 0; k < NEXP; ++k)
    if (slotbase >= offs[k + 1]) e = k + 1;

  const __bf16* W = webf + (size_t)e * D_DIM * D_DIM;
  int lane = threadIdx.x & 31;
  int wv = threadIdx.x >> 5;
  int colbase = colgrp * 256 + wv * 64;
  int m = lane & 15, khalf = lane >> 4, n = m;

  int tk = tok_ids[slotbase + m];
  const __bf16* arow = xbf + (size_t)tk * D_DIM;
  const __bf16* brow = W + (size_t)(colbase + n) * D_DIM + khalf * 16;

  v8f c0 = {}, c1 = {}, c2 = {}, c3 = {};
  for (int kb = 0; kb < D_DIM; kb += 32) {
    v16bf a = loadA_pre(arow, kb, khalf);
    v16bf b0 = *(const v16bf*)(brow + kb);
    v16bf b1 = *(const v16bf*)(brow + (size_t)16 * D_DIM + kb);
    v16bf b2 = *(const v16bf*)(brow + (size_t)32 * D_DIM + kb);
    v16bf b3 = *(const v16bf*)(brow + (size_t)48 * D_DIM + kb);
    c0 = __builtin_amdgcn_wmma_f32_16x16x32_bf16(false, a, false, b0, (short)0, c0, false, false);
    c1 = __builtin_amdgcn_wmma_f32_16x16x32_bf16(false, a, false, b1, (short)0, c1, false, false);
    c2 = __builtin_amdgcn_wmma_f32_16x16x32_bf16(false, a, false, b2, (short)0, c2, false, false);
    c3 = __builtin_amdgcn_wmma_f32_16x16x32_bf16(false, a, false, b3, (short)0, c3, false, false);
  }
#pragma unroll
  for (int j = 0; j < 8; ++j) {
    int mj = j + khalf * 8;
    int tj = tok_ids[slotbase + mj];
    float wj = tok_w[slotbase + mj];
    float* o = out + (size_t)tj * D_DIM + colbase + n;
    atomicAdd(o + 0,  c0[j] * wj);
    atomicAdd(o + 16, c1[j] * wj);
    atomicAdd(o + 32, c2[j] * wj);
    atomicAdd(o + 48, c3[j] * wj);
  }
}

// ---------------- 7a. shared GEMM, on-the-fly conversion (fallback) ----------------
__global__ __launch_bounds__(128) void shared_gemm_kernel(
    const float* __restrict__ feat, const float* __restrict__ shared_w,
    float* __restrict__ out) {
  int tile = blockIdx.x >> 4;
  int colgrp = blockIdx.x & 15;
  int lane = threadIdx.x & 31;
  int wv = threadIdx.x >> 5;
  int colbase = colgrp * 64 + wv * 16;
  int m = lane & 15, khalf = lane >> 4, n = m;

  const float* arow = feat + (size_t)(tile * 16 + m) * D_DIM;
  const float* b0 = shared_w + (size_t)(colbase + n) * D_DIM + khalf * 16;
  const float* b1 = shared_w + (size_t)(D_DIM + colbase + n) * D_DIM + khalf * 16;

  v8f c = {};
  for (int kb = 0; kb < D_DIM; kb += 32) {
    v16bf a = loadA_bf(arow, kb, khalf);
    v16bf b = load16_bf_sum(b0 + kb, b1 + kb);
    c = __builtin_amdgcn_wmma_f32_16x16x32_bf16(false, a, false, b, (short)0, c, false, false);
  }
#pragma unroll
  for (int j = 0; j < 8; ++j) {
    int mj = j + khalf * 8;
    out[(size_t)(tile * 16 + mj) * D_DIM + colbase + n] = c[j];
  }
}

// ---------------- 7b. grouped expert GEMM, on-the-fly conversion (fallback) ----------------
__global__ __launch_bounds__(128) void moe_gemm_kernel(
    const float* __restrict__ feat, const float* __restrict__ expert_w,
    const int* __restrict__ tok_ids, const float* __restrict__ tok_w,
    const int* __restrict__ offs, const int* __restrict__ total_tiles,
    float* __restrict__ out) {
  int tile = blockIdx.x >> 4;
  if (tile >= *total_tiles) return;
  int colgrp = blockIdx.x & 15;
  int slotbase = tile * 16;

  int e = 0;
#pragma unroll
  for (int k = 0; k < NEXP; ++k)
    if (slotbase >= offs[k + 1]) e = k + 1;

  const float* W = expert_w + (size_t)e * D_DIM * D_DIM;
  int lane = threadIdx.x & 31;
  int wv = threadIdx.x >> 5;
  int colbase = colgrp * 64 + wv * 16;
  int m = lane & 15, khalf = lane >> 4, n = m;

  int tk = tok_ids[slotbase + m];
  const float* arow = feat + (size_t)tk * D_DIM;
  const float* brow = W + (size_t)(colbase + n) * D_DIM + khalf * 16;

  v8f c = {};
  for (int kb = 0; kb < D_DIM; kb += 32) {
    v16bf a = loadA_bf(arow, kb, khalf);
    v16bf b = load16_bf(brow + kb);
    c = __builtin_amdgcn_wmma_f32_16x16x32_bf16(false, a, false, b, (short)0, c, false, false);
  }
#pragma unroll
  for (int j = 0; j < 8; ++j) {
    int mj = j + khalf * 8;
    int tj = tok_ids[slotbase + mj];
    float wj = tok_w[slotbase + mj];
    atomicAdd(&out[(size_t)tj * D_DIM + colbase + n], c[j] * wj);
  }
}

extern "C" void kernel_launch(void* const* d_in, const int* in_sizes, int n_in,
                              void* d_out, int out_size, void* d_ws, size_t ws_size,
                              hipStream_t stream) {
  const float* feat     = (const float*)d_in[0];
  const float* router_w = (const float*)d_in[1];
  const float* shared_w = (const float*)d_in[2];
  const float* expert_w = (const float*)d_in[3];
  float* out = (float*)d_out;
  float* aux = out + (size_t)T_TOK * D_DIM;

  char* w = (char*)d_ws;
  int*   cnt         = (int*)(w + 0);
  int*   cursor      = (int*)(w + 64);
  int*   offs        = (int*)(w + 128);            // 17 ints
  int*   total_tiles = (int*)(w + 256);
  float* pi_sum      = (float*)(w + 320);
  int*   topk_idx    = (int*)(w + 512);                          // T*2 ints
  float* topk_wt     = (float*)(w + 512 + 65536);                // T*2 floats
  int*   tok_ids     = (int*)(w + 512 + 131072);                 // MAX_SLOTS ints
  float* tok_w       = (float*)(w + 512 + 131072 + 66560);       // MAX_SLOTS floats

  const size_t BF_BASE    = 264704;                      // end of control region
  const size_t FEAT_BF_SZ = (size_t)T_TOK * D_DIM * 2;   // 16 MB
  const size_t SHRD_BF_SZ = (size_t)D_DIM * D_DIM * 2;   //  2 MB
  const size_t EXPT_BF_SZ = (size_t)NEXP * D_DIM * D_DIM * 2; // 32 MB
  const size_t NEED = BF_BASE + FEAT_BF_SZ + SHRD_BF_SZ + EXPT_BF_SZ;

  moe_init_kernel<<<65, 256, 0, stream>>>(cnt, cursor, pi_sum, tok_ids, tok_w);
  router_kernel<<<T_TOK / 256, 256, 0, stream>>>(feat, router_w, topk_idx, topk_wt,
                                                 cnt, pi_sum);
  scan_kernel<<<1, 32, 0, stream>>>(cnt, pi_sum, offs, total_tiles, aux);
  scatter_kernel<<<T_TOK / 256, 256, 0, stream>>>(topk_idx, topk_wt, offs, cursor,
                                                  tok_ids, tok_w);

  if (ws_size >= NEED) {
    __bf16* feat_bf = (__bf16*)(w + BF_BASE);
    __bf16* shrd_bf = (__bf16*)(w + BF_BASE + FEAT_BF_SZ);
    __bf16* expt_bf = (__bf16*)(w + BF_BASE + FEAT_BF_SZ + SHRD_BF_SZ);

    int feat_n4 = (T_TOK * D_DIM) / 4;          // 2097152
    int shrd_n4 = (D_DIM * D_DIM) / 4;          //  262144
    int expt_n4 = (NEXP * D_DIM * D_DIM) / 4;   // 4194304

    cvt_f32_bf16_kernel<<<feat_n4 / 256, 256, 0, stream>>>(feat, feat_bf, feat_n4);
    cvt_shared_combine_kernel<<<shrd_n4 / 256, 256, 0, stream>>>(shared_w, shrd_bf, shrd_n4);
    cvt_f32_bf16_kernel<<<expt_n4 / 256, 256, 0, stream>>>(expert_w, expt_bf, expt_n4);

    shared_gemm_bf_kernel<<<(T_TOK / 16) * 4, 128, 0, stream>>>(feat_bf, shrd_bf, out);
    moe_gemm_bf_kernel<<<MAX_TILES * 4, 128, 0, stream>>>(feat_bf, expt_bf, tok_ids,
                                                          tok_w, offs, total_tiles, out);
  } else {
    shared_gemm_kernel<<<(T_TOK / 16) * 16, 128, 0, stream>>>(feat, shared_w, out);
    moe_gemm_kernel<<<MAX_TILES * 16, 128, 0, stream>>>(feat, expert_w, tok_ids, tok_w,
                                                        offs, total_tiles, out);
  }
}
